// GRUContextNetworkJAX_87024627352061
// MI455X (gfx1250) — compile-verified, matrix operand order checked by reading, not run on hardware
//
#include <hip/hip_runtime.h>
#include <cstdint>
#include <cmath>

typedef __bf16 bf16_t;
typedef __attribute__((ext_vector_type(16))) __bf16 v16bf;
typedef __attribute__((ext_vector_type(8)))  __bf16 v8bf;
typedef __attribute__((ext_vector_type(8)))  float  v8f;
typedef __attribute__((ext_vector_type(4)))  float  v4f;

#define B_  64
#define T_  512
#define D_  512
#define H_  512
#define G3H (3 * H_)

// workspace layout (bytes)
#define OFF_WIT   0                                   // [3H, D] bf16, transposed Wi
#define OFF_WHT   ((size_t)G3H * D_ * 2)              // [3H, D] bf16, transposed Wh
#define OFF_HPING (2 * (size_t)G3H * D_ * 2)          // [B, H] f32
#define OFF_HPONG (OFF_HPING + (size_t)B_ * H_ * 4)   // [B, H] f32
#define OFF_HBFA  (OFF_HPONG + (size_t)B_ * H_ * 4)   // [B, H] bf16 shadow
#define OFF_HBFB  (OFF_HBFA  + (size_t)B_ * H_ * 2)   // [B, H] bf16 shadow

// ---------------------------------------------------------------------------
// Prep: transpose+convert Wi/Wh (f32 [D,3H] -> bf16 [3H,D]); zero h0 (f32+bf16)
// ---------------------------------------------------------------------------
__global__ void gru_prep_kernel(const float* __restrict__ Wi,
                                const float* __restrict__ Wh,
                                bf16_t* __restrict__ WiT,
                                bf16_t* __restrict__ WhT,
                                float*  __restrict__ h0f,
                                bf16_t* __restrict__ h0bf)
{
    int idx = blockIdx.x * blockDim.x + threadIdx.x;   // col*D + d
    if (idx < G3H * D_) {
        int col = idx >> 9;          // 0..1535
        int d   = idx & (D_ - 1);    // 0..511
        WiT[idx] = (bf16_t)Wi[(size_t)d * G3H + col];
        WhT[idx] = (bf16_t)Wh[(size_t)d * G3H + col];
        if (idx < B_ * H_) { h0f[idx] = 0.0f; h0bf[idx] = (bf16_t)0.0f; }
    }
}

// ---------------------------------------------------------------------------
// One GRU timestep, fully fused, split-K:
//   Grid: 32 blocks x 256 threads (8 waves). Block owns 16 gate-columns.
//   Wave w: M-tile = w&3 (16 batch rows), K-half = w>>2 (256 of K=512).
//   h (bf16 shadow) is staged to LDS via GLOBAL_LOAD_ASYNC_TO_LDS_B128;
//   after the K-loops the same 64KB LDS is reused for the split-K reduction.
// ---------------------------------------------------------------------------
__global__ void __launch_bounds__(256)
gru_step_kernel(const float*  __restrict__ x,
                const bf16_t* __restrict__ WiT,
                const bf16_t* __restrict__ WhT,
                const float*  __restrict__ bi,
                const float*  __restrict__ bhn,
                const float*  __restrict__ h_in,
                const bf16_t* __restrict__ hbf_in,
                float*        __restrict__ h_out,
                bf16_t*       __restrict__ hbf_out,
                float*        __restrict__ out,
                int t)
{
    __shared__ char smem[B_ * H_ * 2];                 // 64 KB, dual-purpose
    bf16_t* hs  = (bf16_t*)smem;                       // phase 1: h as bf16
    float*  red = (float*)smem;                        // phase 2: split-K scratch

    const int tid = threadIdx.x;

    // ---- async-copy h (bf16, 64KB) from global into LDS (ASYNCcnt path) ----
    {
        const uint32_t lds_lo = (uint32_t)(uintptr_t)(void*)smem;
        #pragma unroll
        for (int it = 0; it < 16; ++it) {
            uint32_t off = (uint32_t)tid * 16u + (uint32_t)it * 4096u;
            asm volatile("global_load_async_to_lds_b128 %0, %1, %2"
                         :: "v"(lds_lo + off), "v"(off), "s"(hbf_in)
                         : "memory");
        }
        asm volatile("s_wait_asynccnt 0" ::: "memory");
    }
    __syncthreads();

    const int lane  = tid & 31;
    const int wave  = tid >> 5;
    const int mt    = wave & 3;        // M-tile (batch rows mt*16..+15)
    const int kh    = wave >> 2;       // K-half: 0 -> K[0,256), 1 -> K[256,512)
    const int lhalf = lane >> 4;
    const int l15   = lane & 15;
    const int mrow  = mt * 16 + l15;   // A-matrix row owned by this lane

    const int col16 = blockIdx.x * 16;
    const int ncol  = col16 + l15;     // B-matrix column (gate space, 0..511)

    const float*  xrow = x + ((size_t)mrow * T_ + t) * D_;
    const bf16_t* hrow = hs + mrow * H_;

    // prefetch next timestep's x row (HBM-streamed operand)
    if (t + 1 < T_)
        __builtin_prefetch(x + ((size_t)mrow * T_ + (t + 1)) * D_, 0, 1);

    const bf16_t* wir = WiT + (size_t)(ncol          ) * D_;
    const bf16_t* wiz = WiT + (size_t)(ncol +     H_ ) * D_;
    const bf16_t* win = WiT + (size_t)(ncol + 2 * H_ ) * D_;
    const bf16_t* whr = WhT + (size_t)(ncol          ) * D_;
    const bf16_t* whz = WhT + (size_t)(ncol +     H_ ) * D_;
    const bf16_t* whn = WhT + (size_t)(ncol + 2 * H_ ) * D_;

    v8f air = {}, aiz = {}, ain = {}, ahr = {}, ahz = {}, ahn = {};

    const int kbase = kh * (D_ / 2);
    #pragma unroll 4
    for (int kk = 0; kk < D_ / 2; kk += 32) {
        const int k0 = kbase + kk + lhalf * 8;

        // A (x_t): f32 global -> bf16 fragment
        v4f x0 = *(const v4f*)(xrow + k0);
        v4f x1 = *(const v4f*)(xrow + k0 + 4);
        v4f x2 = *(const v4f*)(xrow + k0 + 16);
        v4f x3 = *(const v4f*)(xrow + k0 + 20);
        v16bf ax;
        #pragma unroll
        for (int e = 0; e < 4; ++e) {
            ax[e]      = (bf16_t)x0[e];
            ax[4 + e]  = (bf16_t)x1[e];
            ax[8 + e]  = (bf16_t)x2[e];
            ax[12 + e] = (bf16_t)x3[e];
        }

        // A (h): bf16 fragment from LDS (two ds_load_b128)
        v8bf hv0 = *(const v8bf*)(hrow + k0);
        v8bf hv1 = *(const v8bf*)(hrow + k0 + 16);
        v16bf ah;
        #pragma unroll
        for (int e = 0; e < 8; ++e) { ah[e] = hv0[e]; ah[8 + e] = hv1[e]; }

        // B fragments: two contiguous 16B loads from transposed weights
        auto ldb = [&](const bf16_t* p) -> v16bf {
            v8bf b0 = *(const v8bf*)(p + k0);
            v8bf b1 = *(const v8bf*)(p + k0 + 16);
            v16bf r;
            #pragma unroll
            for (int e = 0; e < 8; ++e) { r[e] = b0[e]; r[8 + e] = b1[e]; }
            return r;
        };

        air = __builtin_amdgcn_wmma_f32_16x16x32_bf16(false, ax, false, ldb(wir), (short)0, air, false, false);
        aiz = __builtin_amdgcn_wmma_f32_16x16x32_bf16(false, ax, false, ldb(wiz), (short)0, aiz, false, false);
        ain = __builtin_amdgcn_wmma_f32_16x16x32_bf16(false, ax, false, ldb(win), (short)0, ain, false, false);
        ahr = __builtin_amdgcn_wmma_f32_16x16x32_bf16(false, ah, false, ldb(whr), (short)0, ahr, false, false);
        ahz = __builtin_amdgcn_wmma_f32_16x16x32_bf16(false, ah, false, ldb(whz), (short)0, ahz, false, false);
        ahn = __builtin_amdgcn_wmma_f32_16x16x32_bf16(false, ah, false, ldb(whn), (short)0, ahn, false, false);
    }

    // ---- split-K reduction: K-half 1 publishes partials through LDS ----
    __syncthreads();                    // everyone done reading hs
    if (kh == 1) {
        const int base = ((mt * 6) * 32 + lane) * 8;    // per-gate stride 32*8
        *(v8f*)&red[base + 0 * 256] = air;
        *(v8f*)&red[base + 1 * 256] = aiz;
        *(v8f*)&red[base + 2 * 256] = ain;
        *(v8f*)&red[base + 3 * 256] = ahr;
        *(v8f*)&red[base + 4 * 256] = ahz;
        *(v8f*)&red[base + 5 * 256] = ahn;
    }
    __syncthreads();

    if (kh == 0) {
        const int base = ((mt * 6) * 32 + lane) * 8;
        air += *(const v8f*)&red[base + 0 * 256];
        aiz += *(const v8f*)&red[base + 1 * 256];
        ain += *(const v8f*)&red[base + 2 * 256];
        ahr += *(const v8f*)&red[base + 3 * 256];
        ahz += *(const v8f*)&red[base + 4 * 256];
        ahn += *(const v8f*)&red[base + 5 * 256];

        // Gate math + state update (fp32). C/D layout: VGPR e -> row e+8*lhalf,
        // column = lane&15.
        const int   j   = ncol;
        const float bir = bi[j];
        const float biz = bi[H_ + j];
        const float bin = bi[2 * H_ + j];
        const float bn  = bhn[j];

        #pragma unroll
        for (int e = 0; e < 8; ++e) {
            const int brow = mt * 16 + e + 8 * lhalf;
            float r  = 1.0f / (1.0f + __expf(-(air[e] + bir + ahr[e])));
            float z  = 1.0f / (1.0f + __expf(-(aiz[e] + biz + ahz[e])));
            float nn = tanhf(ain[e] + bin + r * (ahn[e] + bn));
            float ho = h_in[brow * H_ + j];
            float hn = (1.0f - z) * nn + z * ho;
            h_out[brow * H_ + j]   = hn;
            hbf_out[brow * H_ + j] = (bf16_t)hn;              // bf16 shadow
            out[((size_t)brow * T_ + t) * H_ + j] = hn;
            if (t == T_ - 1)
                out[(size_t)B_ * T_ * H_ + (size_t)brow * H_ + j] = hn;  // carry
        }
    }
}

// ---------------------------------------------------------------------------
extern "C" void kernel_launch(void* const* d_in, const int* in_sizes, int n_in,
                              void* d_out, int out_size, void* d_ws, size_t ws_size,
                              hipStream_t stream)
{
    const float* x   = (const float*)d_in[0];
    const float* Wi  = (const float*)d_in[1];
    const float* bi  = (const float*)d_in[2];
    const float* Wh  = (const float*)d_in[3];
    const float* bhn = (const float*)d_in[4];
    float* out = (float*)d_out;

    char* ws = (char*)d_ws;
    bf16_t* WiT  = (bf16_t*)(ws + OFF_WIT);
    bf16_t* WhT  = (bf16_t*)(ws + OFF_WHT);
    float*  hA   = (float*)(ws + OFF_HPING);
    float*  hB   = (float*)(ws + OFF_HPONG);
    bf16_t* hbfA = (bf16_t*)(ws + OFF_HBFA);
    bf16_t* hbfB = (bf16_t*)(ws + OFF_HBFB);

    // 1) convert/transpose weights, zero initial carry (f32 + bf16 shadow)
    gru_prep_kernel<<<(G3H * D_ + 255) / 256, 256, 0, stream>>>(Wi, Wh, WiT, WhT, hA, hbfA);

    // 2) 512 dependent timesteps; h ping-pongs between two ws buffers
    for (int t = 0; t < T_; ++t) {
        const float*  hin   = (t & 1) ? hB : hA;
        float*        hout  = (t & 1) ? hA : hB;
        const bf16_t* hbfi  = (t & 1) ? hbfB : hbfA;
        bf16_t*       hbfo  = (t & 1) ? hbfA : hbfB;
        gru_step_kernel<<<32, 256, 0, stream>>>(x, WiT, WhT, bi, bhn,
                                                hin, hbfi, hout, hbfo, out, t);
    }
}
